// GraphUnet_8933531976315
// MI455X (gfx1250) — compile-verified, matrix operand order checked by reading, not run on hardware
//
#include <hip/hip_runtime.h>
#include <hip/hip_bf16.h>

#define NN   4096
#define DD   128
#define EE   131072
#define KK   2457
#define KP   2560

typedef __attribute__((ext_vector_type(16))) __bf16 v16bf;
typedef __attribute__((ext_vector_type(8)))  float  v8f;
typedef __attribute__((ext_vector_type(8)))  int    v8i;

__device__ __forceinline__ unsigned short f2bf(float f) {
  unsigned int u = __float_as_uint(f);
  unsigned int r = (u + 0x7FFFu + ((u >> 16) & 1u)) >> 16;
  return (unsigned short)r;
}

__device__ __forceinline__ void wait_async0() {
#if __has_builtin(__builtin_amdgcn_s_wait_asynccnt)
  __builtin_amdgcn_s_wait_asynccnt(0);
#else
  asm volatile("s_wait_asynccnt 0" ::: "memory");
#endif
}
__device__ __forceinline__ void wait_async8() {
#if __has_builtin(__builtin_amdgcn_s_wait_asynccnt)
  __builtin_amdgcn_s_wait_asynccnt(8);
#else
  asm volatile("s_wait_asynccnt 8" ::: "memory");
#endif
}

__device__ __forceinline__ void async_g2l_b128(unsigned lds_off, const void* gaddr) {
  asm volatile("global_load_async_to_lds_b128 %0, %1, off"
               :: "v"(lds_off), "v"(gaddr) : "memory");
}

__device__ __forceinline__ v16bf ldfrag_bf(const unsigned short* M, int ld, int r0, int k0) {
  int lane = threadIdx.x & 31;
  int r = r0 + (lane & 15);
  int h = lane >> 4;
  const unsigned short* p = M + (size_t)r * ld + k0 + 8 * h;
  union { v16bf v; unsigned int u[8]; } f;
#pragma unroll
  for (int i = 0; i < 4; ++i) {
    f.u[i]     = *(const unsigned int*)(p + 2 * i);
    f.u[i + 4] = *(const unsigned int*)(p + 16 + 2 * i);
  }
  return f.v;
}

__device__ __forceinline__ v8i ldfrag_u8(const unsigned char* M, int ld, int r0, int k0) {
  int lane = threadIdx.x & 31;
  int r = r0 + (lane & 15);
  int h = lane >> 4;
  const unsigned char* p = M + (size_t)r * ld + k0 + 8 * h;
  union { v8i v; unsigned long long q[4]; } f;
#pragma unroll
  for (int i = 0; i < 4; ++i)
    f.q[i] = *(const unsigned long long*)(p + 16 * i);
  return f.v;
}

__global__ void k_zero_f32(float* p, long long n) {
  long long t = (long long)blockIdx.x * blockDim.x + threadIdx.x;
  long long s = (long long)gridDim.x * blockDim.x;
  for (; t < n; t += s) p[t] = 0.f;
}

__global__ void k_scatter(const int* g, float* adj) {
  int e = blockIdx.x * blockDim.x + threadIdx.x;
  if (e < EE) {
    int s = g[e], d = g[EE + e];
    atomicAdd(&adj[(size_t)s * NN + d], 1.0f);
  }
}

__global__ void k_ung(const float* adj, unsigned char* ung) {
  long long t = (long long)blockIdx.x * blockDim.x + threadIdx.x;
  long long s = (long long)gridDim.x * blockDim.x;
  long long n = (long long)NN * NN;
  for (; t < n; t += s) ung[t] = adj[t] > 0.f ? 1 : 0;
}

__global__ void k_tobf(const float* src, unsigned short* dst, int n) {
  int t = blockIdx.x * blockDim.x + threadIdx.x;
  if (t < n) dst[t] = f2bf(src[t]);
}

__global__ void k_transpose_bf(const float* W, unsigned short* WT) {
  int t = blockIdx.x * blockDim.x + threadIdx.x;
  if (t < DD * DD) {
    int nn = t / DD, kk = t % DD;
    WT[t] = f2bf(W[kk * DD + nn]);
  }
}

__global__ void k_gemm_bf16(const unsigned short* A, int lda,
                            const unsigned short* BT, int ldb,
                            const float* bias, float* C, int ldc,
                            int tilesM, int Kdim) {
  int wave = blockIdx.x * (blockDim.x >> 5) + (threadIdx.x >> 5);
  int lane = threadIdx.x & 31;
  if (wave >= tilesM * 2) return;
  int tm = (wave >> 2) * 2, tn = (wave & 3) * 2;
  v8f a00 = {}, a01 = {}, a10 = {}, a11 = {};
  for (int k0 = 0; k0 < Kdim; k0 += 32) {
    if (k0 + 256 < Kdim) {
      __builtin_prefetch(A + (size_t)(tm * 16 + (lane & 15)) * lda + k0 + 256, 0, 3);
      __builtin_prefetch(BT + (size_t)(tn * 16 + (lane & 15)) * ldb + k0 + 256, 0, 3);
    }
    v16bf fa0 = ldfrag_bf(A, lda, tm * 16, k0);
    v16bf fa1 = ldfrag_bf(A, lda, tm * 16 + 16, k0);
    v16bf fb0 = ldfrag_bf(BT, ldb, tn * 16, k0);
    v16bf fb1 = ldfrag_bf(BT, ldb, tn * 16 + 16, k0);
    a00 = __builtin_amdgcn_wmma_f32_16x16x32_bf16(false, fa0, false, fb0, (short)0, a00, false, false);
    a01 = __builtin_amdgcn_wmma_f32_16x16x32_bf16(false, fa0, false, fb1, (short)0, a01, false, false);
    a10 = __builtin_amdgcn_wmma_f32_16x16x32_bf16(false, fa1, false, fb0, (short)0, a10, false, false);
    a11 = __builtin_amdgcn_wmma_f32_16x16x32_bf16(false, fa1, false, fb1, (short)0, a11, false, false);
  }
  int h = lane >> 4;
#pragma unroll
  for (int bi = 0; bi < 2; ++bi) {
#pragma unroll
    for (int bj = 0; bj < 2; ++bj) {
      const v8f& acc = bi == 0 ? (bj == 0 ? a00 : a01) : (bj == 0 ? a10 : a11);
      int nn = (tn + bj) * 16 + (lane & 15);
      float bv = bias ? bias[nn] : 0.f;
#pragma unroll
      for (int r = 0; r < 8; ++r) {
        int row = (tm + bi) * 16 + r + 8 * h;
        C[(size_t)row * ldc + nn] = acc[r] + bv;
      }
    }
  }
}

__global__ void k_score(const float* h0, const float* pw, const float* pb, float* scores) {
  int row = blockIdx.x * (blockDim.x >> 5) + (threadIdx.x >> 5);
  int lane = threadIdx.x & 31;
  if (row >= NN) return;
  float s = 0.f;
  for (int c = lane; c < DD; c += 32) s += h0[(size_t)row * DD + c] * pw[c];
#pragma unroll
  for (int o = 16; o > 0; o >>= 1) s += __shfl_down(s, o, 32);
  if (lane == 0) scores[row] = 1.f / (1.f + __expf(-(s + pb[0])));
}

__global__ void k_topk(const float* scores, int* idx, float* vals) {
  __shared__ float ss[NN];
  __shared__ int   si[NN];
  for (int i = threadIdx.x; i < NN; i += blockDim.x) { ss[i] = scores[i]; si[i] = i; }
  __syncthreads();
  for (int k = 2; k <= NN; k <<= 1) {
    for (int j = k >> 1; j > 0; j >>= 1) {
      for (int i = threadIdx.x; i < NN; i += blockDim.x) {
        int x = i ^ j;
        if (x > i) {
          float sa = ss[i], sb = ss[x];
          int ia = si[i], ib = si[x];
          bool aFirst = (sa > sb) || (sa == sb && ia < ib);
          bool up = ((i & k) == 0);
          if (up ? !aFirst : aFirst) {
            ss[i] = sb; ss[x] = sa; si[i] = ib; si[x] = ia;
          }
        }
      }
      __syncthreads();
    }
  }
  for (int i = threadIdx.x; i < NN; i += blockDim.x) { idx[i] = si[i]; vals[i] = ss[i]; }
}

__global__ void k_newh(const float* h0, const int* idx, const float* vals,
                       unsigned short* newh) {
  int t = blockIdx.x * blockDim.x + threadIdx.x;
  if (t >= KP * DD) return;
  int i = t / DD, c = t % DD;
  float v = 0.f;
  if (i < KK) v = h0[(size_t)idx[i] * DD + c] * vals[i];
  newh[t] = f2bf(v);
}

__global__ void k_buildun(const unsigned char* ung, const int* idx,
                          unsigned char* unA, unsigned char* unBT) {
  long long t = (long long)blockIdx.x * blockDim.x + threadIdx.x;
  long long s = (long long)gridDim.x * blockDim.x;
  long long n = (long long)KP * NN;
  for (; t < n; t += s) {
    int i = (int)(t / NN), r = (int)(t % NN);
    unsigned char a = 0, b = 0;
    if (i < KK) {
      int p = idx[i];
      a = ung[(size_t)p * NN + r];
      b = ung[(size_t)r * NN + p];
    }
    unA[t] = a; unBT[t] = b;
  }
}

__global__ void __launch_bounds__(128) k_twohop(const unsigned char* unA,
                                                const unsigned char* unBT,
                                                unsigned char* sub) {
  __shared__ unsigned char As[2][128 * 64];
  __shared__ unsigned char Bs[2][128 * 64];
  const int T = KP / 128;
  int bi = blockIdx.x / T, bj = blockIdx.x % T;
  int tid = threadIdx.x;
  int lane = tid & 31;
  int wv = tid >> 5;
  int wy = (wv >> 1) * 64, wx = (wv & 1) * 64;
  v8i acc[4][4] = {};

  auto stage = [&](int kk, int buf) {
#pragma unroll
    for (int q = 0; q < 4; ++q) {
      int s = q * 128 + tid;
      int row = s >> 2, seg = (s & 3) * 16;
      const unsigned char* ga = unA  + (size_t)(bi * 128 + row) * NN + kk + seg;
      const unsigned char* gb = unBT + (size_t)(bj * 128 + row) * NN + kk + seg;
      async_g2l_b128((unsigned)(size_t)&As[buf][row * 64 + seg], ga);
      async_g2l_b128((unsigned)(size_t)&Bs[buf][row * 64 + seg], gb);
    }
  };

  const int STEPS = NN / 64;
  stage(0, 0);
  for (int k = 0; k < STEPS; ++k) {
    int cur = k & 1;
    if (k + 1 < STEPS) {
      stage((k + 1) * 64, cur ^ 1);
      wait_async8();
    } else {
      wait_async0();
    }
    __syncthreads();

    v8i af[4], bf[4];
#pragma unroll
    for (int i = 0; i < 4; ++i) af[i] = ldfrag_u8(As[cur], 64, wy + i * 16, 0);
#pragma unroll
    for (int j = 0; j < 4; ++j) bf[j] = ldfrag_u8(Bs[cur], 64, wx + j * 16, 0);
#pragma unroll
    for (int i = 0; i < 4; ++i)
#pragma unroll
      for (int j = 0; j < 4; ++j)
        acc[i][j] = __builtin_amdgcn_wmma_i32_16x16x64_iu8(false, af[i], false, bf[j],
                                                           acc[i][j], false, false);
    __syncthreads();
  }

  int h = lane >> 4;
#pragma unroll
  for (int i = 0; i < 4; ++i) {
#pragma unroll
    for (int j = 0; j < 4; ++j) {
      int col = bj * 128 + wx + j * 16 + (lane & 15);
#pragma unroll
      for (int r = 0; r < 8; ++r) {
        int row = bi * 128 + wy + i * 16 + r + 8 * h;
        sub[(size_t)row * KP + col] = (acc[i][j][r] > 0) ? 1 : 0;
      }
    }
  }
}

__global__ void k_deg(const unsigned char* sub, float* deg) {
  int row = blockIdx.x * (blockDim.x >> 5) + (threadIdx.x >> 5);
  int lane = threadIdx.x & 31;
  if (row >= KP) return;
  int s = 0;
  for (int j = lane; j < KP; j += 32) s += sub[(size_t)row * KP + j];
#pragma unroll
  for (int o = 16; o > 0; o >>= 1) s += __shfl_down(s, o, 32);
  if (lane == 0) deg[row] = (float)s;
}

__global__ void k_subbf(const unsigned char* sub, unsigned short* subbf) {
  long long t = (long long)blockIdx.x * blockDim.x + threadIdx.x;
  long long s = (long long)gridDim.x * blockDim.x;
  long long n = (long long)KP * KP;
  for (; t < n; t += s) subbf[t] = sub[t] ? (unsigned short)0x3F80 : (unsigned short)0;
}

__global__ void k_yt(const float* Y, const float* deg, unsigned short* YT) {
  int t = blockIdx.x * blockDim.x + threadIdx.x;
  if (t >= KP * DD) return;
  int j = t / DD, d = t % DD;
  float v = 0.f;
  if (j < KK) { float dg = deg[j]; v = (dg != 0.f) ? Y[(size_t)j * DD + d] / dg : 0.f; }
  YT[(size_t)d * KP + j] = f2bf(v);
}

__global__ void k_hbbf(const float* hb, unsigned short* hbbf) {
  int t = blockIdx.x * blockDim.x + threadIdx.x;
  if (t >= KP * DD) return;
  int i = t / DD;
  hbbf[t] = (i < KK) ? f2bf(hb[t]) : (unsigned short)0;
}

__global__ void k_zt(const float* Z, unsigned short* ZT) {
  int t = blockIdx.x * blockDim.x + threadIdx.x;
  if (t >= KP * DD) return;
  int i = t / DD, d = t % DD;
  float v = (i < KK) ? Z[(size_t)i * DD + d] : 0.f;
  ZT[(size_t)d * KP + i] = f2bf(v);
}

__global__ void k_ap(const float* adj, const int* idx, unsigned short* Ap) {
  long long t = (long long)blockIdx.x * blockDim.x + threadIdx.x;
  long long s = (long long)gridDim.x * blockDim.x;
  long long n = (long long)NN * KP;
  for (; t < n; t += s) {
    int r = (int)(t / KP), i = (int)(t % KP);
    float v = (i < KK) ? adj[(size_t)r * NN + idx[i]] : 0.f;
    Ap[t] = f2bf(v);
  }
}

__global__ void k_final(const float* hup, const float* h0, const float* h, float* out) {
  int t = blockIdx.x * blockDim.x + threadIdx.x;
  if (t >= NN * DD) return;
  float h1 = hup[t] + h0[t];
  out[t] = h1;
  out[NN * DD + t] = h1 + h[t];
}

extern "C" void kernel_launch(void* const* d_in, const int* in_sizes, int n_in,
                              void* d_out, int out_size, void* d_ws, size_t ws_size,
                              hipStream_t stream) {
  (void)in_sizes; (void)n_in; (void)out_size; (void)ws_size;
  const int*   g  = (const int*)d_in[0];
  const float* h  = (const float*)d_in[1];
  const float* Wd = (const float*)d_in[2];
  const float* bd = (const float*)d_in[3];
  const float* pw = (const float*)d_in[4];
  const float* pb = (const float*)d_in[5];
  const float* Wb = (const float*)d_in[6];
  const float* bb = (const float*)d_in[7];
  const float* Wu = (const float*)d_in[8];
  const float* bu = (const float*)d_in[9];
  float* out = (float*)d_out;

  char* w = (char*)d_ws;
  size_t off = 0;
  auto take = [&](size_t bytes) -> char* {
    char* p = w + off;
    off += (bytes + 255) & ~(size_t)255;
    return p;
  };
  float*          adj   = (float*)         take((size_t)NN * NN * 4);
  unsigned char*  ung   = (unsigned char*) take((size_t)NN * NN);
  unsigned short* hbf   = (unsigned short*)take((size_t)NN * DD * 2);
  unsigned short* WdT   = (unsigned short*)take((size_t)DD * DD * 2);
  unsigned short* WbT   = (unsigned short*)take((size_t)DD * DD * 2);
  unsigned short* WuT   = (unsigned short*)take((size_t)DD * DD * 2);
  float*          h0    = (float*)         take((size_t)NN * DD * 4);
  float*          sc    = (float*)         take((size_t)NN * 4);
  int*            idx   = (int*)           take((size_t)NN * 4);
  float*          vals  = (float*)         take((size_t)NN * 4);
  unsigned short* newh  = (unsigned short*)take((size_t)KP * DD * 2);
  unsigned char*  unA   = (unsigned char*) take((size_t)KP * NN);
  unsigned char*  unBT  = (unsigned char*) take((size_t)KP * NN);
  unsigned char*  sub   = (unsigned char*) take((size_t)KP * KP);
  unsigned short* subbf = (unsigned short*)take((size_t)KP * KP * 2);
  float*          deg   = (float*)         take((size_t)KP * 4);
  float*          Yf    = (float*)         take((size_t)KP * DD * 4);
  unsigned short* YT    = (unsigned short*)take((size_t)DD * KP * 2);
  float*          hb    = (float*)         take((size_t)KP * DD * 4);
  unsigned short* hbbf  = (unsigned short*)take((size_t)KP * DD * 2);
  float*          Zf    = (float*)         take((size_t)KP * DD * 4);
  unsigned short* ZT    = (unsigned short*)take((size_t)DD * KP * 2);
  unsigned short* Ap    = (unsigned short*)take((size_t)NN * KP * 2);
  float*          hup   = (float*)         take((size_t)NN * DD * 4);

  const int TB = 256;
  k_zero_f32<<<16384, TB, 0, stream>>>(adj, (long long)NN * NN);
  k_scatter<<<(EE + TB - 1) / TB, TB, 0, stream>>>(g, adj);
  k_ung<<<16384, TB, 0, stream>>>(adj, ung);
  k_tobf<<<(NN * DD + TB - 1) / TB, TB, 0, stream>>>(h, hbf, NN * DD);
  k_transpose_bf<<<(DD * DD + TB - 1) / TB, TB, 0, stream>>>(Wd, WdT);
  k_transpose_bf<<<(DD * DD + TB - 1) / TB, TB, 0, stream>>>(Wb, WbT);
  k_transpose_bf<<<(DD * DD + TB - 1) / TB, TB, 0, stream>>>(Wu, WuT);
  k_gemm_bf16<<<(NN / 16) * 2 / 4, 128, 0, stream>>>(hbf, DD, WdT, DD, bd, h0, DD, NN / 16, DD);
  k_score<<<NN / 4, 128, 0, stream>>>(h0, pw, pb, sc);
  k_topk<<<1, 1024, 0, stream>>>(sc, idx, vals);
  k_newh<<<(KP * DD + TB - 1) / TB, TB, 0, stream>>>(h0, idx, vals, newh);
  k_buildun<<<16384, TB, 0, stream>>>(ung, idx, unA, unBT);
  k_twohop<<<(KP / 128) * (KP / 128), 128, 0, stream>>>(unA, unBT, sub);
  k_deg<<<(KP + 3) / 4, 128, 0, stream>>>(sub, deg);
  k_subbf<<<16384, TB, 0, stream>>>(sub, subbf);
  k_gemm_bf16<<<(KP / 16) * 2 / 4, 128, 0, stream>>>(newh, DD, WbT, DD, nullptr, Yf, DD, KP / 16, DD);
  k_yt<<<(KP * DD + TB - 1) / TB, TB, 0, stream>>>(Yf, deg, YT);
  k_gemm_bf16<<<(KP / 16) * 2 / 4, 128, 0, stream>>>(subbf, KP, YT, KP, bb, hb, DD, KP / 16, KP);
  k_hbbf<<<(KP * DD + TB - 1) / TB, TB, 0, stream>>>(hb, hbbf);
  k_gemm_bf16<<<(KP / 16) * 2 / 4, 128, 0, stream>>>(hbbf, DD, WuT, DD, nullptr, Zf, DD, KP / 16, DD);
  k_zt<<<(KP * DD + TB - 1) / TB, TB, 0, stream>>>(Zf, ZT);
  k_ap<<<16384, TB, 0, stream>>>(adj, idx, Ap);
  k_gemm_bf16<<<(NN / 16) * 2 / 4, 128, 0, stream>>>(Ap, KP, ZT, KP, bu, hup, DD, NN / 16, KP);
  k_final<<<(NN * DD + TB - 1) / TB, TB, 0, stream>>>(hup, h0, h, out);
}
